// CVRPModel_24421184045097
// MI455X (gfx1250) — matrix operand order repaired and checked by reading, not compiled, not run on hardware
//
#include <hip/hip_runtime.h>
#include <hip/hip_bf16.h>
#include <cstdint>

typedef __attribute__((ext_vector_type(16))) _Float16 v16h;
typedef __attribute__((ext_vector_type(8)))  float    v8f;
typedef unsigned int v4u  __attribute__((ext_vector_type(4)));
typedef int          v4i_ __attribute__((ext_vector_type(4)));
typedef int          v8i_ __attribute__((ext_vector_type(8)));

#define BATCH 128
#define NNODE 200
#define DMODEL 128
#define NHEAD 8
#define NLAYER 3
#define FFDIM 2048
#define MROWS (BATCH * NNODE)            // 25600
#define MAX_STEPS (2 * (NNODE - 1) + 1)  // 399

__device__ __forceinline__ v8f wmma32(v16h a, v16h b, v8f c) {
    return __builtin_amdgcn_wmma_f32_16x16x32_f16(
        /*neg_a=*/false, a, /*neg_b=*/false, b,
        /*c_mod=*/(short)0, c, /*reuse_a=*/false, /*reuse_b=*/false);
}

// TDM: DMA a 16-row x cols-elem f16 tile (row stride = stride_elems) into LDS.
// Issue from one wave only; TDM ignores EXEC. Descriptor per CDNA5 ISA D# layout.
__device__ __forceinline__ void tdm_load_tile16(unsigned lds_off, const void* gaddr,
                                                unsigned rows, unsigned cols,
                                                unsigned stride_elems) {
    unsigned long long ga = (unsigned long long)(uintptr_t)gaddr;
    v4u g0;
    g0[0] = 1u;                                            // count=1 (valid user D#)
    g0[1] = lds_off;                                       // lds_addr (bytes)
    g0[2] = (unsigned)(ga & 0xFFFFFFFFu);                  // global_addr[31:0]
    g0[3] = (unsigned)((ga >> 32) & 0x1FFFFFFu) | (2u << 30); // addr[56:32] | type=2
    v8i_ g1;
    g1[0] = (int)(1u << 16);                               // wg_mask=0, data_size=1 (2B)
    g1[1] = (int)((cols & 0xFFFFu) << 16);                 // tensor_dim0[15:0]
    g1[2] = (int)(((cols >> 16) & 0xFFFFu) | ((rows & 0xFFFFu) << 16)); // dim0 hi | dim1 lo
    g1[3] = (int)(((rows >> 16) & 0xFFFFu) | ((cols & 0xFFFFu) << 16)); // dim1 hi | tile_dim0
    g1[4] = (int)16u;                                      // tile_dim1=16, tile_dim2=0
    g1[5] = (int)stride_elems;                             // tensor_dim0_stride[31:0]
    g1[6] = 0;
    g1[7] = 0;
    v4i_ z4 = {0, 0, 0, 0};
#if __clang_major__ >= 23
    v8i_ z8 = {0, 0, 0, 0, 0, 0, 0, 0};
    __builtin_amdgcn_tensor_load_to_lds(g0, g1, z4, z4, z8, 0);
#else
    __builtin_amdgcn_tensor_load_to_lds(g0, g1, z4, z4, 0);
#endif
}

// ---------------------------------------------------------------- convert
__global__ void f32_to_f16_kernel(const float* __restrict__ src,
                                  _Float16* __restrict__ dst, int n) {
    int i = blockIdx.x * blockDim.x + threadIdx.x;
    if (i < n) dst[i] = (_Float16)src[i];
}

// ---------------------------------------------------------------- embed
__global__ void embed_kernel(const float* __restrict__ coords,
                             const float* __restrict__ dem,
                             const float* __restrict__ cap,
                             const float* __restrict__ eW,
                             const float* __restrict__ eb,
                             float* __restrict__ h, _Float16* __restrict__ h16) {
    int idx = blockIdx.x * blockDim.x + threadIdx.x;   // over B*N*D
    if (idx >= MROWS * DMODEL) return;
    int d = idx & (DMODEL - 1);
    int bn = idx >> 7;
    int b = bn / NNODE;
    float ratio = dem[bn] / cap[b];
    float x0 = coords[2 * bn], x1 = coords[2 * bn + 1];
    float v = eW[d * 3] * x0 + eW[d * 3 + 1] * x1 + eW[d * 3 + 2] * ratio + eb[d];
    h[idx] = v;
    h16[idx] = (_Float16)v;
}

// ---------------------------------------------------------------- generic WMMA GEMM
// C[M,N] = A16[M,128] * W16[N,128]^T + bias ; optional relu. K fixed at 128 so
// the fragment array indexes are compile-time constants (no MOVREL indexing).
// A tile staged to LDS once per block via TDM; A-fragments live in registers
// across all column tiles.
__launch_bounds__(256)
__global__ void wmma_gemm_kernel(const _Float16* __restrict__ A,
                                 const _Float16* __restrict__ W,
                                 const float* __restrict__ bias,
                                 float* __restrict__ Cf,
                                 int M, int N, int relu) {
    constexpr int K = DMODEL;            // 128
    __shared__ _Float16 sA[16 * K];
    const int w    = threadIdx.x >> 5;
    const int lane = threadIdx.x & 31;
    const int half = lane >> 4;
    const int l16  = lane & 15;
    const int row0 = blockIdx.x * 16;

    if (threadIdx.x < 32) {   // wave 0 issues the tensor DMA
        tdm_load_tile16((unsigned)(uintptr_t)sA, A + (size_t)row0 * K,
                        (unsigned)M, (unsigned)K, (unsigned)K);
        __builtin_amdgcn_s_wait_tensorcnt(0);
    }
    __syncthreads();

    v16h afr[4];
#pragma unroll
    for (int kt = 0; kt < 4; ++kt) {
#pragma unroll
        for (int i = 0; i < 8; ++i) {
            int k = kt * 32 + 16 * half + 2 * i;
            afr[kt][2 * i]     = sA[l16 * K + k];
            afr[kt][2 * i + 1] = sA[l16 * K + k + 1];
        }
    }
    const int ntiles = N >> 4;
    for (int ct = w; ct < ntiles; ct += 8) {
        const int col = ct * 16 + l16;
        if (ct + 8 < ntiles)
            __builtin_prefetch(W + (size_t)(col + 128) * K, 0, 3);
        v8f c;
        float bv = bias ? bias[col] : 0.0f;
#pragma unroll
        for (int r = 0; r < 8; ++r) c[r] = bv;
#pragma unroll
        for (int kt = 0; kt < 4; ++kt) {
            v16h bf;
#pragma unroll
            for (int i = 0; i < 8; ++i) {
                int kb = kt * 32 + 16 * half + 2 * i;
                bf[2 * i]     = W[(size_t)col * K + kb];
                bf[2 * i + 1] = W[(size_t)col * K + kb + 1];
            }
            c = wmma32(afr[kt], bf, c);
        }
#pragma unroll
        for (int r = 0; r < 8; ++r) {
            int row = row0 + r + 8 * half;
            float v = c[r];
            if (relu) v = fmaxf(v, 0.0f);
            Cf[(size_t)row * N + col] = v;
        }
    }
}

// ---------------------------------------------------------------- attention (per b,head)
__launch_bounds__(256)
__global__ void attention_kernel(const float* __restrict__ qkv,
                                 _Float16* __restrict__ attn16) {
    const int b  = blockIdx.x / NHEAD;
    const int hh = blockIdx.x % NHEAD;
    __shared__ float Ks[NNODE * 16];
    __shared__ float Vs[NNODE * 16];
    const int t = threadIdx.x;
    for (int idx = t; idx < NNODE * 16; idx += 256) {
        int n = idx >> 4, d = idx & 15;
        size_t base = ((size_t)(b * NNODE + n)) * (3 * DMODEL) + hh * 16 + d;
        Ks[idx] = qkv[base + DMODEL];
        Vs[idx] = qkv[base + 2 * DMODEL];
    }
    __syncthreads();
    const int w = t >> 5, lane = t & 31;
    for (int row = w; row < NNODE; row += 8) {
        float q[16];
        const float* qp = qkv + ((size_t)(b * NNODE + row)) * (3 * DMODEL) + hh * 16;
#pragma unroll
        for (int d = 0; d < 16; ++d) q[d] = qp[d];
        float sv[7];
        int cnt = 0;
        float mx = -1e30f;
        for (int j = lane; j < NNODE; j += 32) {
            const float* kp = Ks + j * 16;
            float s = 0.f;
#pragma unroll
            for (int d = 0; d < 16; ++d) s += q[d] * kp[d];
            s *= 0.25f;                      // 1/sqrt(dh), dh=16
            sv[cnt++] = s;
            mx = fmaxf(mx, s);
        }
#pragma unroll
        for (int msk = 16; msk >= 1; msk >>= 1) mx = fmaxf(mx, __shfl_xor(mx, msk, 32));
        float sum = 0.f, acc[16];
#pragma unroll
        for (int d = 0; d < 16; ++d) acc[d] = 0.f;
        cnt = 0;
        for (int j = lane; j < NNODE; j += 32) {
            float e = __expf(sv[cnt++] - mx);
            sum += e;
            const float* vp = Vs + j * 16;
#pragma unroll
            for (int d = 0; d < 16; ++d) acc[d] += e * vp[d];
        }
#pragma unroll
        for (int msk = 16; msk >= 1; msk >>= 1) {
            sum += __shfl_xor(sum, msk, 32);
#pragma unroll
            for (int d = 0; d < 16; ++d) acc[d] += __shfl_xor(acc[d], msk, 32);
        }
        if (lane == 0) {
            float inv = 1.0f / sum;
            _Float16* op = attn16 + ((size_t)(b * NNODE + row)) * DMODEL + hh * 16;
#pragma unroll
            for (int d = 0; d < 16; ++d) op[d] = (_Float16)(acc[d] * inv);
        }
    }
}

// ---------------------------------------------------------------- residual + LayerNorm
__launch_bounds__(256)
__global__ void add_ln_kernel(const float* __restrict__ x, const float* __restrict__ r,
                              const float* __restrict__ g, const float* __restrict__ bb,
                              float* __restrict__ out, _Float16* __restrict__ out16, int M) {
    const int w = threadIdx.x >> 5, lane = threadIdx.x & 31;
    const int row = blockIdx.x * 8 + w;
    if (row >= M) return;
    const float* xp = x + (size_t)row * DMODEL;
    const float* rp = r + (size_t)row * DMODEL;
    const int d0 = lane * 4;
    float v[4];
#pragma unroll
    for (int i = 0; i < 4; ++i) v[i] = xp[d0 + i] + rp[d0 + i];
    float s = v[0] + v[1] + v[2] + v[3];
#pragma unroll
    for (int msk = 16; msk >= 1; msk >>= 1) s += __shfl_xor(s, msk, 32);
    float mean = s * (1.0f / DMODEL);
    float vs = 0.f;
#pragma unroll
    for (int i = 0; i < 4; ++i) { float dd = v[i] - mean; vs += dd * dd; }
#pragma unroll
    for (int msk = 16; msk >= 1; msk >>= 1) vs += __shfl_xor(vs, msk, 32);
    float rstd = rsqrtf(vs * (1.0f / DMODEL) + 1e-5f);
#pragma unroll
    for (int i = 0; i < 4; ++i) {
        float o = (v[i] - mean) * rstd * g[d0 + i] + bb[d0 + i];
        out[(size_t)row * DMODEL + d0 + i] = o;
        out16[(size_t)row * DMODEL + d0 + i] = (_Float16)o;
    }
}

// ---------------------------------------------------------------- fused FFN
// GEMM1 (+relu) into LDS, GEMM2 out of LDS. A tile staged via TDM into the
// front of fbuf, register-preloaded, then fbuf reused as the f16 intermediate.
__launch_bounds__(256)
__global__ void ffn_kernel(const _Float16* __restrict__ A,
                           const _Float16* __restrict__ W1, const float* __restrict__ b1,
                           const _Float16* __restrict__ W2, const float* __restrict__ b2,
                           float* __restrict__ out) {
    __shared__ _Float16 fbuf[16 * FFDIM];    // 64 KB
    const int w    = threadIdx.x >> 5;
    const int lane = threadIdx.x & 31;
    const int half = lane >> 4;
    const int l16  = lane & 15;
    const int row0 = blockIdx.x * 16;

    if (threadIdx.x < 32) {
        tdm_load_tile16((unsigned)(uintptr_t)fbuf, A + (size_t)row0 * DMODEL,
                        (unsigned)MROWS, DMODEL, DMODEL);
        __builtin_amdgcn_s_wait_tensorcnt(0);
    }
    __syncthreads();

    v16h afr[4];
#pragma unroll
    for (int kt = 0; kt < 4; ++kt) {
#pragma unroll
        for (int i = 0; i < 8; ++i) {
            int k = kt * 32 + 16 * half + 2 * i;
            afr[kt][2 * i]     = fbuf[l16 * DMODEL + k];
            afr[kt][2 * i + 1] = fbuf[l16 * DMODEL + k + 1];
        }
    }
    __syncthreads();   // all A reads complete before fbuf is overwritten

    // stage 1: f = relu(A @ W1^T + b1), 2048 cols / 8 waves
    for (int ct = w; ct < FFDIM / 16; ct += 8) {
        const int col = ct * 16 + l16;
        if (ct + 8 < FFDIM / 16)
            __builtin_prefetch(W1 + (size_t)(col + 128) * DMODEL, 0, 3);
        v8f c;
        float bv = b1[col];
#pragma unroll
        for (int r = 0; r < 8; ++r) c[r] = bv;
#pragma unroll
        for (int kt = 0; kt < 4; ++kt) {
            v16h bf;
#pragma unroll
            for (int i = 0; i < 8; ++i) {
                int k = kt * 32 + 16 * half + 2 * i;
                bf[2 * i]     = W1[(size_t)col * DMODEL + k];
                bf[2 * i + 1] = W1[(size_t)col * DMODEL + k + 1];
            }
            c = wmma32(afr[kt], bf, c);
        }
#pragma unroll
        for (int r = 0; r < 8; ++r) {
            int rl = r + 8 * half;
            fbuf[rl * FFDIM + col] = (_Float16)fmaxf(c[r], 0.0f);
        }
    }
    __syncthreads();
    // stage 2: out = f @ W2^T + b2 ; 128 cols / 8 waves = 1 tile each, K = 2048
    {
        const int col = w * 16 + l16;
        v8f c;
        float bv = b2[col];
#pragma unroll
        for (int r = 0; r < 8; ++r) c[r] = bv;
        for (int kt = 0; kt < FFDIM / 32; ++kt) {
            int k0 = kt * 32;
            if (k0 + 512 < FFDIM)
                __builtin_prefetch(W2 + (size_t)col * FFDIM + k0 + 512, 0, 3);
            v16h af, bf;
#pragma unroll
            for (int i = 0; i < 8; ++i) {
                int ka = k0 + 16 * half + 2 * i;
                af[2 * i]     = fbuf[l16 * FFDIM + ka];
                af[2 * i + 1] = fbuf[l16 * FFDIM + ka + 1];
                int kb = k0 + 16 * half + 2 * i;
                bf[2 * i]     = W2[(size_t)col * FFDIM + kb];
                bf[2 * i + 1] = W2[(size_t)col * FFDIM + kb + 1];
            }
            c = wmma32(af, bf, c);
        }
#pragma unroll
        for (int r = 0; r < 8; ++r) {
            int row = row0 + r + 8 * half;
            out[(size_t)row * DMODEL + col] = c[r];
        }
    }
}

// ---------------------------------------------------------------- mean pool
__global__ void mean_kernel(const float* __restrict__ h, float* __restrict__ gctx) {
    int b = blockIdx.x, d = threadIdx.x;
    float s = 0.f;
    for (int n = 0; n < NNODE; ++n) s += h[((size_t)(b * NNODE + n)) * DMODEL + d];
    gctx[b * DMODEL + d] = s * (1.0f / NNODE);
}

// ---------------------------------------------------------------- decode
__device__ __forceinline__ float rng_uniform(unsigned long long x) {
    x ^= x >> 33; x *= 0xff51afd7ed558ccdULL;
    x ^= x >> 33; x *= 0xc4ceb9fe1a85ec53ULL;
    x ^= x >> 33;
    unsigned r = (unsigned)(x >> 40);
    return (r + 0.5f) * (1.0f / 16777216.0f);
}

__launch_bounds__(128)
__global__ void decode_kernel(const float* __restrict__ h, const float* __restrict__ gctx,
                              const float* __restrict__ kmat, const float* __restrict__ dem,
                              const float* __restrict__ cap,
                              const float* __restrict__ Wcap, const float* __restrict__ bcap,
                              const float* __restrict__ Wctx, const float* __restrict__ bctx,
                              const float* __restrict__ Wq,
                              float* __restrict__ path_out, float* __restrict__ logp_out) {
    const int b = blockIdx.x, t = threadIdx.x;
    __shared__ float qin[3 * DMODEL], qc[DMODEL], qv[DMODEL], lg[NNODE];
    __shared__ float red[128];
    __shared__ int redi[128];
    __shared__ unsigned char visited[NNODE];
    __shared__ int s_cur, s_done, s_hasfeas, s_notall;
    __shared__ float s_rem, s_lsum;
    for (int n = t; n < NNODE; n += 128) visited[n] = 0;
    if (t == 0) {
        s_cur = 0; s_done = 0; s_rem = 1.0f; s_lsum = 0.0f;
        path_out[(size_t)b * (1 + MAX_STEPS)] = 0.0f;
    }
    __syncthreads();
    const float invcap = 1.0f / cap[b];
    const float scale = 0.08838834764831845f;   // 1/sqrt(128)

    for (int step = 0; step < MAX_STEPS; ++step) {
        const int cur = s_cur;
        const float rem = s_rem;
        const int done = s_done;
        for (int idx = t; idx < 3 * DMODEL; idx += 128) {
            float v;
            if (idx < DMODEL)          v = h[((size_t)(b * NNODE + cur)) * DMODEL + idx];
            else if (idx < 2 * DMODEL) v = gctx[b * DMODEL + (idx - DMODEL)];
            else                       v = Wcap[idx - 2 * DMODEL] * rem + bcap[idx - 2 * DMODEL];
            qin[idx] = v;
        }
        __syncthreads();
        {
            float a = bctx[t];
            const float* wp = Wctx + (size_t)t * (3 * DMODEL);
            for (int j = 0; j < 3 * DMODEL; ++j) a += wp[j] * qin[j];
            qc[t] = a;
        }
        __syncthreads();
        {
            float a = 0.f;
            const float* wp = Wq + (size_t)t * DMODEL;
            for (int j = 0; j < DMODEL; ++j) a += wp[j] * qc[j];
            qv[t] = a;
        }
        __syncthreads();
        if (t == 0) { s_hasfeas = 0; s_notall = 0; }
        __syncthreads();
        for (int n = t; n < NNODE; n += 128) {
            const float* kp = kmat + ((size_t)(b * NNODE + n)) * DMODEL;
            float s = 0.f;
            for (int j = 0; j < DMODEL; ++j) s += kp[j] * qv[j];
            lg[n] = s * scale;
            if (n >= 1) {
                float dr = dem[b * NNODE + n] * invcap;
                if (!visited[n]) {
                    atomicOr(&s_notall, 1);
                    if (dr <= rem) atomicOr(&s_hasfeas, 1);
                }
            }
        }
        __syncthreads();
        const int has_feas   = s_hasfeas;
        const int all_served = !s_notall;
        const int at_depot   = (cur == 0) && !done;
        const int not_depot  = (cur != 0) && !done;
        const int force_depot = (all_served && not_depot) || (not_depot && !has_feas) || done;
        const int depot_mask = (at_depot && has_feas) || (not_depot && has_feas && !all_served);
        for (int n = t; n < NNODE; n += 128) {
            float dr = dem[b * NNODE + n] * invcap;
            int inf = visited[n] || (dr > rem);
            if (n == 0 && depot_mask) inf = 1;
            if (inf) lg[n] = -1e9f;
        }
        __syncthreads();
        float mx = -1e30f, gmx = -1e30f;
        int gmi = 0;
        for (int n = t; n < NNODE; n += 128) {
            float v = lg[n];
            mx = fmaxf(mx, v);
            unsigned long long seed = (((unsigned long long)b) << 40) ^
                                      (((unsigned long long)step) << 20) ^
                                      (unsigned long long)n ^ 0x9E3779B97F4A7C15ULL;
            float u = rng_uniform(seed);
            float pv = v - __logf(-__logf(u));
            if (pv > gmx) { gmx = pv; gmi = n; }
        }
        red[t] = mx; __syncthreads();
        for (int off = 64; off >= 1; off >>= 1) {
            if (t < off) red[t] = fmaxf(red[t], red[t + off]);
            __syncthreads();
        }
        const float rowmax = red[0]; __syncthreads();
        float se = 0.f;
        for (int n = t; n < NNODE; n += 128) se += __expf(lg[n] - rowmax);
        red[t] = se; __syncthreads();
        for (int off = 64; off >= 1; off >>= 1) {
            if (t < off) red[t] += red[t + off];
            __syncthreads();
        }
        const float lse = rowmax + __logf(red[0]); __syncthreads();
        red[t] = gmx; redi[t] = gmi; __syncthreads();
        for (int off = 64; off >= 1; off >>= 1) {
            if (t < off && red[t + off] > red[t]) { red[t] = red[t + off]; redi[t] = redi[t + off]; }
            __syncthreads();
        }
        if (t == 0) {
            int sel = redi[0];
            float slp = lg[sel] - lse;
            if (force_depot) { sel = 0; slp = 0.0f; }
            const int is_depot = (sel == 0);
            float take = dem[b * NNODE + sel] * invcap;
            float nr = is_depot ? 1.0f : (rem - take);
            s_rem = fminf(fmaxf(nr, 0.0f), 1.0f);
            if (!is_depot) visited[sel] = 1;
            if (is_depot && all_served) s_done = 1;
            s_cur = sel;
            s_lsum += slp;
            path_out[(size_t)b * (1 + MAX_STEPS) + 1 + step] = (float)sel;
        }
        __syncthreads();
    }
    if (t == 0) logp_out[b] = s_lsum;
}

// ---------------------------------------------------------------- launch
extern "C" void kernel_launch(void* const* d_in, const int* in_sizes, int n_in,
                              void* d_out, int out_size, void* d_ws, size_t ws_size,
                              hipStream_t stream) {
    const float* coords   = (const float*)d_in[0];
    const float* demands  = (const float*)d_in[1];
    const float* capacity = (const float*)d_in[2];
    const float* embed_W  = (const float*)d_in[3];
    const float* embed_b  = (const float*)d_in[4];
    const float* Wqkv     = (const float*)d_in[5];
    const float* bqkv     = (const float*)d_in[6];
    const float* Wo       = (const float*)d_in[7];
    const float* bo       = (const float*)d_in[8];
    const float* W1       = (const float*)d_in[9];
    const float* b1       = (const float*)d_in[10];
    const float* W2       = (const float*)d_in[11];
    const float* b2       = (const float*)d_in[12];
    const float* ln1_g    = (const float*)d_in[13];
    const float* ln1_b    = (const float*)d_in[14];
    const float* ln2_g    = (const float*)d_in[15];
    const float* ln2_b    = (const float*)d_in[16];
    const float* Wq       = (const float*)d_in[17];
    const float* Wk       = (const float*)d_in[18];
    const float* Wcap     = (const float*)d_in[19];
    const float* bcap     = (const float*)d_in[20];
    const float* Wctx     = (const float*)d_in[21];
    const float* bctx     = (const float*)d_in[22];

    char* ws = (char*)d_ws;
    size_t off = 0;
    auto alloc = [&](size_t bytes) { size_t o = off; off = (off + bytes + 255) & ~(size_t)255; return o; };
    const size_t M = MROWS;
    float*    h_f32  = (float*)   (ws + alloc(M * DMODEL * 4));
    _Float16* h16    = (_Float16*)(ws + alloc(M * DMODEL * 2));
    float*    qkv    = (float*)   (ws + alloc(M * 3 * DMODEL * 4));
    _Float16* attn16 = (_Float16*)(ws + alloc(M * DMODEL * 2));
    float*    tmp    = (float*)   (ws + alloc(M * DMODEL * 4));
    float*    kmat   = (float*)   (ws + alloc(M * DMODEL * 4));
    float*    gctx   = (float*)   (ws + alloc(BATCH * DMODEL * 4));
    _Float16* Wqkv16 = (_Float16*)(ws + alloc((size_t)NLAYER * 3 * DMODEL * DMODEL * 2));
    _Float16* Wo16   = (_Float16*)(ws + alloc((size_t)NLAYER * DMODEL * DMODEL * 2));
    _Float16* W116   = (_Float16*)(ws + alloc((size_t)NLAYER * FFDIM * DMODEL * 2));
    _Float16* W216   = (_Float16*)(ws + alloc((size_t)NLAYER * DMODEL * FFDIM * 2));
    _Float16* Wk16   = (_Float16*)(ws + alloc((size_t)DMODEL * DMODEL * 2));

    auto cvt = [&](const float* s, _Float16* d, int n) {
        f32_to_f16_kernel<<<(n + 255) / 256, 256, 0, stream>>>(s, d, n);
    };
    cvt(Wqkv, Wqkv16, NLAYER * 3 * DMODEL * DMODEL);
    cvt(Wo,   Wo16,   NLAYER * DMODEL * DMODEL);
    cvt(W1,   W116,   NLAYER * FFDIM * DMODEL);
    cvt(W2,   W216,   NLAYER * DMODEL * FFDIM);
    cvt(Wk,   Wk16,   DMODEL * DMODEL);

    embed_kernel<<<(M * DMODEL) / 256, 256, 0, stream>>>(
        coords, demands, capacity, embed_W, embed_b, h_f32, h16);

    for (int i = 0; i < NLAYER; ++i) {
        wmma_gemm_kernel<<<M / 16, 256, 0, stream>>>(
            h16, Wqkv16 + (size_t)i * 3 * DMODEL * DMODEL, bqkv + i * 3 * DMODEL,
            qkv, (int)M, 3 * DMODEL, 0);
        attention_kernel<<<BATCH * NHEAD, 256, 0, stream>>>(qkv, attn16);
        wmma_gemm_kernel<<<M / 16, 256, 0, stream>>>(
            attn16, Wo16 + (size_t)i * DMODEL * DMODEL, bo + i * DMODEL,
            tmp, (int)M, DMODEL, 0);
        add_ln_kernel<<<M / 8, 256, 0, stream>>>(
            tmp, h_f32, ln1_g + i * DMODEL, ln1_b + i * DMODEL, h_f32, h16, (int)M);
        ffn_kernel<<<M / 16, 256, 0, stream>>>(
            h16, W116 + (size_t)i * FFDIM * DMODEL, b1 + i * FFDIM,
            W216 + (size_t)i * DMODEL * FFDIM, b2 + i * DMODEL, tmp);
        add_ln_kernel<<<M / 8, 256, 0, stream>>>(
            tmp, h_f32, ln2_g + i * DMODEL, ln2_b + i * DMODEL, h_f32, h16, (int)M);
    }

    mean_kernel<<<BATCH, DMODEL, 0, stream>>>(h_f32, gctx);
    wmma_gemm_kernel<<<M / 16, 256, 0, stream>>>(
        h16, Wk16, nullptr, kmat, (int)M, DMODEL, 0);

    float* path_out = (float*)d_out;
    float* logp_out = path_out + (size_t)BATCH * (1 + MAX_STEPS);
    decode_kernel<<<BATCH, 128, 0, stream>>>(
        h_f32, gctx, kmat, demands, capacity,
        Wcap, bcap, Wctx, bctx, Wq, path_out, logp_out);
}
